// RWKV_TimeMix_54485955117236
// MI455X (gfx1250) — compile-verified, matrix operand order checked by reading, not run on hardware
//
#include <hip/hip_runtime.h>
#include <hip/hip_bf16.h>

// ---------------------------------------------------------------------------
// RWKV TimeMix for MI455X (gfx1250, wave32, WMMA).
//   k = X*Wk^T ; r = sigmoid(X*Wr^T) ; s_t = w*s_{t-1}+k_t ; out = (r*s)*Wv^T
// GEMMs: v_wmma_f32_16x16x32_bf16, 128x128 block tile, double-buffered LDS,
// async global->LDS copies (ASYNCcnt) when the toolchain exposes the builtin.
// ---------------------------------------------------------------------------

typedef __attribute__((ext_vector_type(16))) __bf16 v16bf;
typedef __attribute__((ext_vector_type(8)))  __bf16 v8bf;
typedef __attribute__((ext_vector_type(8)))  float  v8f;
typedef int v4i __attribute__((vector_size(16)));

#if defined(__AMDGCN__) && __has_builtin(__builtin_amdgcn_global_load_async_to_lds_b128)
#define USE_ASYNC_COPY 1
#else
#define USE_ASYNC_COPY 0
#endif

#if USE_ASYNC_COPY
#if __has_builtin(__builtin_amdgcn_s_wait_asynccnt)
#define ASYNC_WAIT(n) __builtin_amdgcn_s_wait_asynccnt(n)
#else
#define ASYNC_WAIT(n) asm volatile("s_wait_asynccnt %0" ::"i"(n) : "memory")
#endif

__device__ __forceinline__ void async_cp_b128(const __bf16* gsrc, __bf16* ldst)
{
    // addrspacecast first (generic -> AS1 / AS3), then pointee bitcast.
    auto g1 = (const __attribute__((address_space(1))) __bf16*)gsrc;
    auto g  = (__attribute__((address_space(1))) v4i*)g1;
    auto l3 = (__attribute__((address_space(3))) __bf16*)ldst;
    auto l  = (__attribute__((address_space(3))) v4i*)l3;
    __builtin_amdgcn_global_load_async_to_lds_b128(g, l, /*offset=*/0, /*cpol=*/0);
}
#endif

// ---------------------------------------------------------------------------
// fp32 -> bf16 conversion, 4 elements / thread
// ---------------------------------------------------------------------------
__global__ __launch_bounds__(256)
void cvt_f32_to_bf16(const float* __restrict__ in, __bf16* __restrict__ out, int n4)
{
    int i = blockIdx.x * blockDim.x + threadIdx.x;
    if (i < n4) {
        float4 f = ((const float4*)in)[i];
        ushort4 o;
        __bf16 a = (__bf16)f.x, b = (__bf16)f.y, c = (__bf16)f.z, d = (__bf16)f.w;
        o.x = __builtin_bit_cast(unsigned short, a);
        o.y = __builtin_bit_cast(unsigned short, b);
        o.z = __builtin_bit_cast(unsigned short, c);
        o.w = __builtin_bit_cast(unsigned short, d);
        ((ushort4*)out)[i] = o;
    }
}

// ---------------------------------------------------------------------------
// bf16 WMMA GEMM:  Out[m,n] = sum_k A[m,k] * B[n,k]   (Y = A * B^T)
//   Block tile 128(M) x 128(N), K-step 32, 8 waves = 4(M) x 2(N),
//   each wave owns 32x64 = 2x4 wmma 16x16x32 accumulators.
//   LDS rows padded to 40 bf16 (80B): the 16 per-lane b128 fragment reads
//   land on distinct bank groups. Double-buffered, async-copied when possible.
// ---------------------------------------------------------------------------
template<bool SIGMOID, bool OUT_BF16>
__global__ __launch_bounds__(256)
void gemm_bf16_wmma(const __bf16* __restrict__ A, const __bf16* __restrict__ B,
                    void* __restrict__ Out, int Ndim, int Kdim)
{
    constexpr int BM = 128, BN = 128, BK = 32, LDT = 40;
    constexpr int NCH = 2;   // b128 chunks per thread per tile (A and B each)

    __shared__ __attribute__((aligned(16))) __bf16 sA[2][BM * LDT];
    __shared__ __attribute__((aligned(16))) __bf16 sB[2][BN * LDT];

    const int tid = threadIdx.x;
    const int bM  = blockIdx.y * BM;
    const int bN  = blockIdx.x * BN;

    // ---- tile-staging geometry: 128x32 bf16 = 512 b128 chunks, 2/thread ----
    int chRow[NCH], chCol[NCH];
    const __bf16* aBase[NCH];
    const __bf16* bBase[NCH];
#pragma unroll
    for (int t = 0; t < NCH; ++t) {
        int id    = tid + t * 256;
        chRow[t]  = id >> 2;            // 4 chunks per 32-elem row
        chCol[t]  = (id & 3) * 8;
        aBase[t]  = A + (size_t)(bM + chRow[t]) * Kdim + chCol[t];
        bBase[t]  = B + (size_t)(bN + chRow[t]) * Kdim + chCol[t];
    }

    // ---- wave geometry -----------------------------------------------------
    const int wave    = tid >> 5;
    const int lane    = tid & 31;
    const int waveM   = (wave & 3) * 32;
    const int waveN   = (wave >> 2) * 64;
    const int laneRow = lane & 15;
    const int laneHi  = lane >> 4;        // 0: K 0-7/16-23, 1: K 8-15/24-31
    const int kOff    = laneHi ? 8 : 0;

    v8f acc[2][4] = {};

    auto compute_tile = [&](int buf) {
        v16bf aF[2], bF[4];
#pragma unroll
        for (int i = 0; i < 2; ++i) {
            const __bf16* p = &sA[buf][(waveM + i * 16 + laneRow) * LDT + kOff];
            v8bf lo = *(const v8bf*)p;
            v8bf hi = *(const v8bf*)(p + 16);
            aF[i] = __builtin_shufflevector(lo, hi,
                        0,1,2,3,4,5,6,7,8,9,10,11,12,13,14,15);
        }
#pragma unroll
        for (int j = 0; j < 4; ++j) {
            const __bf16* p = &sB[buf][(waveN + j * 16 + laneRow) * LDT + kOff];
            v8bf lo = *(const v8bf*)p;
            v8bf hi = *(const v8bf*)(p + 16);
            bF[j] = __builtin_shufflevector(lo, hi,
                        0,1,2,3,4,5,6,7,8,9,10,11,12,13,14,15);
        }
#pragma unroll
        for (int i = 0; i < 2; ++i)
#pragma unroll
            for (int j = 0; j < 4; ++j)
                acc[i][j] = __builtin_amdgcn_wmma_f32_16x16x32_bf16(
                    false, aF[i], false, bF[j],
                    (short)0, acc[i][j], false, false);
    };

    const int NK = Kdim / BK;

#if USE_ASYNC_COPY
    auto issue_tile = [&](int buf, int k0) {
#pragma unroll
        for (int t = 0; t < NCH; ++t)
            async_cp_b128(aBase[t] + k0, &sA[buf][chRow[t] * LDT + chCol[t]]);
#pragma unroll
        for (int t = 0; t < NCH; ++t)
            async_cp_b128(bBase[t] + k0, &sB[buf][chRow[t] * LDT + chCol[t]]);
    };

    issue_tile(0, 0);
    for (int kt = 0; kt < NK; ++kt) {
        const int cur = kt & 1;
        if (kt + 1 < NK) {
            issue_tile(cur ^ 1, (kt + 1) * BK);
            ASYNC_WAIT(2 * NCH);    // current tile's copies done, next in flight
        } else {
            ASYNC_WAIT(0);
        }
        __syncthreads();            // all waves see this buffer's LDS data
        compute_tile(cur);
        __syncthreads();            // compute done before buffer is re-filled
    }
#else
    // Fallback: global->VGPR staging, single LDS buffer, prefetch next K-tile.
    uint4 aReg[NCH], bReg[NCH];
#pragma unroll
    for (int t = 0; t < NCH; ++t) {
        aReg[t] = *(const uint4*)aBase[t];
        bReg[t] = *(const uint4*)bBase[t];
    }
    for (int kt = 0; kt < NK; ++kt) {
        const int k0 = kt * BK;
#pragma unroll
        for (int t = 0; t < NCH; ++t) {
            *(uint4*)&sA[0][chRow[t] * LDT + chCol[t]] = aReg[t];
            *(uint4*)&sB[0][chRow[t] * LDT + chCol[t]] = bReg[t];
        }
        __syncthreads();
        if (kt + 1 < NK) {
#pragma unroll
            for (int t = 0; t < NCH; ++t) {
                aReg[t] = *(const uint4*)(aBase[t] + k0 + BK);
                bReg[t] = *(const uint4*)(bBase[t] + k0 + BK);
            }
        }
        compute_tile(0);
        __syncthreads();
    }
#endif

    // ---- epilogue: C/D layout = 8 VGPRs, M = v + 8*laneHi, N = laneRow -----
#pragma unroll
    for (int i = 0; i < 2; ++i) {
#pragma unroll
        for (int j = 0; j < 4; ++j) {
#pragma unroll
            for (int v = 0; v < 8; ++v) {
                int row = bM + waveM + i * 16 + v + (laneHi << 3);
                int col = bN + waveN + j * 16 + laneRow;
                float val = acc[i][j][v];
                if (SIGMOID) val = 1.0f / (1.0f + __expf(-val));
                size_t o = (size_t)row * Ndim + col;
                if (OUT_BF16) ((__bf16*)Out)[o] = (__bf16)val;
                else          ((float*)Out)[o] = val;
            }
        }
    }
}

// ---------------------------------------------------------------------------
// Per-channel linear scan: s_t = w*s_{t-1} + k_t, emit rs = r_t * s_t (bf16).
// ---------------------------------------------------------------------------
__global__ __launch_bounds__(256)
void rwkv_scan(const __bf16* __restrict__ Kb, const __bf16* __restrict__ Rb,
               const float* __restrict__ time_decay, __bf16* __restrict__ RS,
               int T, int C)
{
    const int cpb = C / 256;
    const int b   = blockIdx.x / cpb;
    const int c   = (blockIdx.x % cpb) * 256 + threadIdx.x;
    const float w = -__expf(time_decay[c]);
    float s = 0.0f;
    const size_t base = (size_t)b * T * C + c;

    for (int t = 0; t < T; t += 8) {
        float kv[8], rv[8];
#pragma unroll
        for (int i = 0; i < 8; ++i) {
            size_t idx = base + (size_t)(t + i) * C;
            kv[i] = (float)Kb[idx];
            rv[i] = (float)Rb[idx];
        }
#pragma unroll
        for (int i = 0; i < 8; ++i) {
            s = fmaf(s, w, kv[i]);
            RS[base + (size_t)(t + i) * C] = (__bf16)(rv[i] * s);
        }
    }
}

// ---------------------------------------------------------------------------
extern "C" void kernel_launch(void* const* d_in, const int* in_sizes, int n_in,
                              void* d_out, int out_size, void* d_ws, size_t ws_size,
                              hipStream_t stream)
{
    const float* x  = (const float*)d_in[0];
    const float* Wk = (const float*)d_in[1];
    const float* Wv = (const float*)d_in[2];
    const float* Wr = (const float*)d_in[3];
    const float* td = (const float*)d_in[4];
    // d_in[5] = time_first : unused (vanishes in the reference recurrence)

    const int Bn = 8, T = 2048, C = 1024;
    const size_t Mrows = (size_t)Bn * T;            // 16384

    char* ws = (char*)d_ws;
    size_t off = 0;
    auto carve = [&](size_t bytes) {
        void* p = ws + off;
        off += (bytes + 255) & ~(size_t)255;
        return p;
    };
    __bf16* Xb   = (__bf16*)carve(Mrows * C * 2);
    __bf16* Wkb  = (__bf16*)carve((size_t)C * C * 2);
    __bf16* Wvb  = (__bf16*)carve((size_t)C * C * 2);
    __bf16* Wrb  = (__bf16*)carve((size_t)C * C * 2);
    __bf16* Kbuf = (__bf16*)carve(Mrows * C * 2);
    __bf16* Rbuf = (__bf16*)carve(Mrows * C * 2);
    __bf16* RSb  = (__bf16*)carve(Mrows * C * 2);
    (void)ws_size;

    // 1) fp32 -> bf16
    {
        int n4x = (int)(Mrows * C / 4);
        cvt_f32_to_bf16<<<(n4x + 255) / 256, 256, 0, stream>>>(x, Xb, n4x);
        int n4w = C * C / 4;
        cvt_f32_to_bf16<<<(n4w + 255) / 256, 256, 0, stream>>>(Wk, Wkb, n4w);
        cvt_f32_to_bf16<<<(n4w + 255) / 256, 256, 0, stream>>>(Wv, Wvb, n4w);
        cvt_f32_to_bf16<<<(n4w + 255) / 256, 256, 0, stream>>>(Wr, Wrb, n4w);
    }

    dim3 grid(C / 128, (unsigned)(Mrows / 128));    // (8, 128)
    dim3 block(256);

    // 2) k = X*Wk^T (bf16 out),  r = sigmoid(X*Wr^T) (bf16 out)
    gemm_bf16_wmma<false, true><<<grid, block, 0, stream>>>(Xb, Wkb, Kbuf, C, C);
    gemm_bf16_wmma<true,  true><<<grid, block, 0, stream>>>(Xb, Wrb, Rbuf, C, C);

    // 3) recurrence + r*s fused, emits bf16 RS
    rwkv_scan<<<Bn * (C / 256), 256, 0, stream>>>(Kbuf, Rbuf, td, RSb, T, C);

    // 4) out = RS*Wv^T (fp32 straight to d_out)
    gemm_bf16_wmma<false, false><<<grid, block, 0, stream>>>(RSb, Wvb, d_out, C, C);
}
